// EGNN_17368847745209
// MI455X (gfx1250) — compile-verified, matrix operand order checked by reading, not run on hardware
//
#include <hip/hip_runtime.h>

typedef __attribute__((ext_vector_type(16))) _Float16 v16h;
typedef __attribute__((ext_vector_type(8)))  float    v8f;

// ---------------- problem dims (fixed by reference) ----------------
constexpr int Bsz = 2, Npt = 2048, DIM = 128, KNN = 32, MDIM = 64;
constexpr int EIN  = 2 * DIM + 2;   // 258
constexpr int EINP = 288;           // pad to 9*32
constexpr int H1   = 2 * EIN;       // 516
constexpr int H1P  = 544;           // pad to 17*32
constexpr int HM   = 4 * MDIM;      // 256
constexpr int NIN  = DIM + MDIM;    // 192
constexpr int NH   = 2 * DIM;       // 256
constexpr int NPB  = 2;             // nodes per edge block
constexpr int ER   = NPB * KNN;     // 64 edge rows per block
constexpr int ERT  = ER / 16;       // 4 row tiles

// ---------------- f16 workspace layout (element offsets) ----------------
constexpr size_t OFF_WE1T = 0;                                  // [H1P][EINP]
constexpr size_t OFF_WE2T = OFF_WE1T + (size_t)H1P * EINP;      // [64][H1P]
constexpr size_t OFF_WM1T = OFF_WE2T + (size_t)MDIM * H1P;      // [256][64]
constexpr size_t OFF_WM2T = OFF_WM1T + (size_t)HM * MDIM;       // [16][256]
constexpr size_t OFF_WV1T = OFF_WM2T + (size_t)16 * HM;
constexpr size_t OFF_WV2T = OFF_WV1T + (size_t)HM * MDIM;
constexpr size_t OFF_WN1T = OFF_WV2T + (size_t)16 * HM;         // [256][192]
constexpr size_t OFF_WN2T = OFF_WN1T + (size_t)NH * NIN;        // [128][256]
constexpr size_t OFF_BE1  = OFF_WN2T + (size_t)DIM * NH;
constexpr size_t OFF_BE2  = OFF_BE1 + H1P;
constexpr size_t OFF_BM1  = OFF_BE2 + MDIM;
constexpr size_t OFF_BM2  = OFF_BM1 + HM;
constexpr size_t OFF_BV1  = OFF_BM2 + 16;
constexpr size_t OFF_BV2  = OFF_BV1 + HM;
constexpr size_t OFF_BN1  = OFF_BV2 + 16;
constexpr size_t OFF_BN2  = OFF_BN1 + NH;
constexpr size_t OFF_FEATSH = OFF_BN2 + DIM;
constexpr size_t HALF_TOTAL = OFF_FEATSH + (size_t)Bsz * Npt * DIM;

// ---------------- WMMA fragment helpers (ISA 7.12.2 layouts) ----------------
// A (16x32 f16): lane l holds row (l&15); k-half select by (l>>4): e<8 -> K=kh*8+e,
// e>=8 -> K=16+kh*8+(e-8).  B stored transposed [N][Kpad] uses the same loader
// with "row" = output column.
__device__ __forceinline__ v16h frag_ld(const _Float16* p, int stride) {
  const int l = threadIdx.x & 31;
  const _Float16* q = p + (l & 15) * stride + ((l >> 4) << 3);
  v16h f;
#pragma unroll
  for (int e = 0; e < 8; ++e) { f[e] = q[e]; f[8 + e] = q[16 + e]; }
  return f;
}

// Column-strip GEMM: one wave owns column tile `colBase`, loads each B fragment
// once per k-step and reuses it across NROWT row-tile accumulators.
template <int KSTEPS, int NROWT>
__device__ __forceinline__ void gemm_colstrip(const _Float16* A, int strideA,
                                              const _Float16* Bt, int strideB,
                                              int colBase, float biasv, v8f* acc) {
#pragma unroll
  for (int rt = 0; rt < NROWT; ++rt)
#pragma unroll
    for (int r = 0; r < 8; ++r) acc[rt][r] = biasv;
#pragma unroll
  for (int ks = 0; ks < KSTEPS; ++ks) {
    v16h bm = frag_ld(Bt + colBase * strideB + ks * 32, strideB);
#pragma unroll
    for (int rt = 0; rt < NROWT; ++rt) {
      v16h a = frag_ld(A + (rt * 16) * strideA + ks * 32, strideA);
      acc[rt] = __builtin_amdgcn_wmma_f32_16x16x32_f16(false, a, false, bm,
                                                       (short)0, acc[rt], false, false);
    }
  }
}

// C/D layout: VGPR r of lane l -> row r + 8*(l>>4), col (l&15)
__device__ __forceinline__ void store_silu(v8f acc, _Float16* dst, int stride,
                                           int rowBase, int colBase) {
  const int l   = threadIdx.x & 31;
  const int col = colBase + (l & 15);
  const int r0  = rowBase + ((l >> 4) << 3);
#pragma unroll
  for (int r = 0; r < 8; ++r) {
    float x = acc[r];
    dst[(r0 + r) * stride + col] = (_Float16)(x / (1.0f + __expf(-x)));
  }
}

// ---------------- weight conversion: f32 [Kin][Nout] -> f16 transposed+padded ----------------
__device__ __forceinline__ void cvt_wT(const float* __restrict__ W, _Float16* __restrict__ Wt,
                                       int Kin, int Nout, int Kpad, int Npad, int tid, int nth) {
  const int total = Npad * Kpad;
  for (int i = tid; i < total; i += nth) {
    const int n = i / Kpad, k = i - n * Kpad;
    const float v = (k < Kin && n < Nout) ? W[k * Nout + n] : 0.0f;
    Wt[i] = (_Float16)v;
  }
}
__device__ __forceinline__ void cvt_b(const float* __restrict__ b, _Float16* __restrict__ bp,
                                      int Nout, int Npad, int tid, int nth) {
  for (int i = tid; i < Npad; i += nth) bp[i] = (_Float16)(i < Nout ? b[i] : 0.0f);
}

__global__ void egnn_convert(const float* __restrict__ feats,
                             const float* We1, const float* be1,
                             const float* We2, const float* be2,
                             const float* Wm1, const float* bm1,
                             const float* Wm2, const float* bm2,
                             const float* Wv1, const float* bv1,
                             const float* Wv2, const float* bv2,
                             const float* Wn1, const float* bn1,
                             const float* Wn2, const float* bn2,
                             _Float16* __restrict__ wsh) {
  const int tid = blockIdx.x * blockDim.x + threadIdx.x;
  const int nth = gridDim.x * blockDim.x;
  cvt_wT(We1, wsh + OFF_WE1T, EIN, H1,  EINP, H1P,  tid, nth);
  cvt_wT(We2, wsh + OFF_WE2T, H1,  MDIM, H1P, MDIM, tid, nth);
  cvt_wT(Wm1, wsh + OFF_WM1T, MDIM, HM, MDIM, HM,   tid, nth);
  cvt_wT(Wm2, wsh + OFF_WM2T, HM,  1,   HM,   16,   tid, nth);
  cvt_wT(Wv1, wsh + OFF_WV1T, MDIM, HM, MDIM, HM,   tid, nth);
  cvt_wT(Wv2, wsh + OFF_WV2T, HM,  1,   HM,   16,   tid, nth);
  cvt_wT(Wn1, wsh + OFF_WN1T, NIN, NH,  NIN,  NH,   tid, nth);
  cvt_wT(Wn2, wsh + OFF_WN2T, NH,  DIM, NH,   DIM,  tid, nth);
  cvt_b(be1, wsh + OFF_BE1, H1,  H1P,  tid, nth);
  cvt_b(be2, wsh + OFF_BE2, MDIM, MDIM, tid, nth);
  cvt_b(bm1, wsh + OFF_BM1, HM, HM, tid, nth);
  cvt_b(bm2, wsh + OFF_BM2, 1, 16, tid, nth);
  cvt_b(bv1, wsh + OFF_BV1, HM, HM, tid, nth);
  cvt_b(bv2, wsh + OFF_BV2, 1, 16, tid, nth);
  cvt_b(bn1, wsh + OFF_BN1, NH, NH, tid, nth);
  cvt_b(bn2, wsh + OFF_BN2, DIM, DIM, tid, nth);
  for (size_t i = tid; i < (size_t)Bsz * Npt * DIM; i += nth)
    wsh[OFF_FEATSH + i] = (_Float16)feats[i];
}

// ---------------- kNN by dist_mean: one block per (b, i) ----------------
__global__ void __launch_bounds__(256) egnn_knn(const float* __restrict__ cmean,
                                                const float* __restrict__ cvar,
                                                int* __restrict__ idxbuf) {
  const int bi  = blockIdx.x;            // bb*Npt + i
  const int bb  = bi / Npt;
  const int tid = threadIdx.x;
  __shared__ float sd[Npt];
  __shared__ float rv[256];
  __shared__ int   ri[256];

  const float mi0 = cmean[bi * 3 + 0], mi1 = cmean[bi * 3 + 1], mi2 = cmean[bi * 3 + 2];
  const float vi0 = cvar[bi * 3 + 0],  vi1 = cvar[bi * 3 + 1],  vi2 = cvar[bi * 3 + 2];

  for (int j = tid; j < Npt; j += 256) {
    const int gj = bb * Npt + j;
    const float r0 = mi0 - cmean[gj * 3 + 0];
    const float r1 = mi1 - cmean[gj * 3 + 1];
    const float r2 = mi2 - cmean[gj * 3 + 2];
    const float w0 = vi0 + cvar[gj * 3 + 0];
    const float w1 = vi1 + cvar[gj * 3 + 1];
    const float w2 = vi2 + cvar[gj * 3 + 2];
    sd[j] = r0 * r0 + r1 * r1 + r2 * r2 + w0 + w1 + w2;
  }
  __syncthreads();

  for (int k = 0; k < KNN; ++k) {
    float best = 3.4e38f; int bj = 0;
    for (int j = tid; j < Npt; j += 256)
      if (sd[j] < best) { best = sd[j]; bj = j; }
    rv[tid] = best; ri[tid] = bj;
    __syncthreads();
    for (int s = 128; s > 0; s >>= 1) {
      if (tid < s && rv[tid + s] < rv[tid]) { rv[tid] = rv[tid + s]; ri[tid] = ri[tid + s]; }
      __syncthreads();
    }
    if (tid == 0) { idxbuf[bi * KNN + k] = ri[0]; sd[ri[0]] = 3.4e38f; }
    __syncthreads();
  }
}

// ---------------- fused edge pipeline: one block (8 waves) per NPB nodes ----------------
__global__ void __launch_bounds__(256) egnn_edge(const float* __restrict__ cmean,
                                                 const float* __restrict__ cvar,
                                                 const _Float16* __restrict__ wsh,
                                                 const int* __restrict__ idxbuf,
                                                 float* __restrict__ mibuf,
                                                 float* __restrict__ cm_out,
                                                 float* __restrict__ cv_out) {
  const int bi0 = blockIdx.x * NPB;      // first node (bb*Npt + i) of this block
  const int tid = threadIdx.x;
  const int wv  = tid >> 5;

  const _Float16* We1t = wsh + OFF_WE1T;
  const _Float16* We2t = wsh + OFF_WE2T;
  const _Float16* Wm1t = wsh + OFF_WM1T;
  const _Float16* Wm2t = wsh + OFF_WM2T;
  const _Float16* Wv1t = wsh + OFF_WV1T;
  const _Float16* Wv2t = wsh + OFF_WV2T;
  const _Float16* be1p = wsh + OFF_BE1;
  const _Float16* be2p = wsh + OFF_BE2;
  const _Float16* bm1p = wsh + OFF_BM1;
  const _Float16* bm2p = wsh + OFF_BM2;
  const _Float16* bv1p = wsh + OFF_BV1;
  const _Float16* bv2p = wsh + OFF_BV2;
  const _Float16* featsh = wsh + OFF_FEATSH;

  // edge_in buffer is dead after layer1; reuse the same LDS as the head-hidden buffer
  __shared__ _Float16 s_inhid[ER * EINP];     // 36.9 KB (>= ER*HM = 32.8 KB)
  __shared__ _Float16 s_act1[ER * H1P];       // 69.6 KB
  __shared__ _Float16 s_mij[ER * MDIM];       // 8.2 KB
  __shared__ float s_dm[ER], s_ds[ER], s_relm[ER][3], s_relv[ER][3];
  __shared__ int   s_j[ER];
  __shared__ float s_wm[ER], s_wv[ER];
  __shared__ float s_red[ER][6];
  _Float16* s_in  = s_inhid;
  _Float16* s_hid = s_inhid;

  // neighbor geometry (recomputed from coords; cheaper than storing dm/ds)
  if (tid < ER) {
    const int bi = bi0 + (tid >> 5);     // owning node
    const int bb = bi / Npt;
    const int j  = idxbuf[bi * KNN + (tid & 31)];
    const int gj = bb * Npt + j;
    s_j[tid] = j;
    float dm = 0.f, tr = 0.f, dq = 0.f;
#pragma unroll
    for (int c = 0; c < 3; ++c) {
      const float rm = cmean[bi * 3 + c] - cmean[gj * 3 + c];
      const float rw = cvar[bi * 3 + c] + cvar[gj * 3 + c];
      s_relm[tid][c] = rm; s_relv[tid][c] = rw;
      dm += rm * rm; tr += rw; dq += rm * rm * rw;
    }
    s_dm[tid] = dm + tr;
    s_ds[tid] = 2.f * tr + 4.f * dq;
  }
  __syncthreads();

  // assemble edge_in [ER x 288] = [feats_i | feats_j | dm | ds | 0...]
  for (int t = tid; t < ER * EINP; t += 256) {
    const int r = t / EINP, c = t - r * EINP;
    const int bi = bi0 + (r >> 5);
    const int bb = bi / Npt;
    _Float16 v = (_Float16)0.0f;
    if      (c < DIM)          v = featsh[(size_t)bi * DIM + c];
    else if (c < 2 * DIM)      v = featsh[((size_t)bb * Npt + s_j[r]) * DIM + (c - DIM)];
    else if (c == 2 * DIM)     v = (_Float16)s_dm[r];
    else if (c == 2 * DIM + 1) v = (_Float16)s_ds[r];
    s_in[t] = v;
  }
  __syncthreads();

  // edge layer 1: [ERx288] @ [288x544], SiLU   (col-strip: B frag reused 4x)
  for (int ct = wv; ct < H1P / 16; ct += 8) {
    const float bv = (float)be1p[ct * 16 + (tid & 15)];
    v8f acc[ERT];
    gemm_colstrip<EINP / 32, ERT>(s_in, EINP, We1t, EINP, ct * 16, bv, acc);
#pragma unroll
    for (int rt = 0; rt < ERT; ++rt) store_silu(acc[rt], s_act1, H1P, rt * 16, ct * 16);
  }
  __syncthreads();

  // edge layer 2: [ERx544] @ [544x64], SiLU -> m_ij
  for (int ct = wv; ct < MDIM / 16; ct += 8) {
    const float bv = (float)be2p[ct * 16 + (tid & 15)];
    v8f acc[ERT];
    gemm_colstrip<H1P / 32, ERT>(s_act1, H1P, We2t, H1P, ct * 16, bv, acc);
#pragma unroll
    for (int rt = 0; rt < ERT; ++rt) store_silu(acc[rt], s_mij, MDIM, rt * 16, ct * 16);
  }
  __syncthreads();

  // mean-head hidden: [ERx64] @ [64x256], SiLU
  for (int ct = wv; ct < HM / 16; ct += 8) {
    const float bv = (float)bm1p[ct * 16 + (tid & 15)];
    v8f acc[ERT];
    gemm_colstrip<MDIM / 32, ERT>(s_mij, MDIM, Wm1t, MDIM, ct * 16, bv, acc);
#pragma unroll
    for (int rt = 0; rt < ERT; ++rt) store_silu(acc[rt], s_hid, HM, rt * 16, ct * 16);
  }
  __syncthreads();
  // mean-head out: [ERx256] @ [256x1(pad16)] -> w_mean[ER]
  if (wv == 0) {
    const float bv = (float)bm2p[tid & 15];
    v8f acc[ERT];
    gemm_colstrip<HM / 32, ERT>(s_hid, HM, Wm2t, HM, 0, bv, acc);
    const int l = tid & 31;
    if ((l & 15) == 0) {
#pragma unroll
      for (int rt = 0; rt < ERT; ++rt) {
        const int r0 = rt * 16 + ((l >> 4) << 3);
#pragma unroll
        for (int r = 0; r < 8; ++r) s_wm[r0 + r] = acc[rt][r];
      }
    }
  }
  __syncthreads();

  // var-head hidden (reuses s_hid after barrier)
  for (int ct = wv; ct < HM / 16; ct += 8) {
    const float bv = (float)bv1p[ct * 16 + (tid & 15)];
    v8f acc[ERT];
    gemm_colstrip<MDIM / 32, ERT>(s_mij, MDIM, Wv1t, MDIM, ct * 16, bv, acc);
#pragma unroll
    for (int rt = 0; rt < ERT; ++rt) store_silu(acc[rt], s_hid, HM, rt * 16, ct * 16);
  }
  __syncthreads();
  if (wv == 0) {
    const float bv = (float)bv2p[tid & 15];
    v8f acc[ERT];
    gemm_colstrip<HM / 32, ERT>(s_hid, HM, Wv2t, HM, 0, bv, acc);
    const int l = tid & 31;
    if ((l & 15) == 0) {
#pragma unroll
      for (int rt = 0; rt < ERT; ++rt) {
        const int r0 = rt * 16 + ((l >> 4) << 3);
#pragma unroll
        for (int r = 0; r < 8; ++r) s_wv[r0 + r] = acc[rt][r];
      }
    }
  }
  __syncthreads();

  // m_i = sum_k m_ij  (per node, 64 cols)
  if (tid < NPB * MDIM) {
    const int nn = tid >> 6, c = tid & 63;
    float s = 0.f;
    for (int r = 0; r < KNN; ++r) s += (float)s_mij[(nn * KNN + r) * MDIM + c];
    mibuf[((size_t)bi0 + nn) * MDIM + c] = s;
  }
  // coordinate mean/var updates
  if (tid < ER) {
#pragma unroll
    for (int c = 0; c < 3; ++c) {
      s_red[tid][c]     = s_wm[tid] * s_relm[tid][c];
      s_red[tid][3 + c] = s_wv[tid] * s_relv[tid][c];
    }
  }
  __syncthreads();
  if (tid < NPB * 6) {
    const int nn = tid / 6, comp = tid % 6;
    float s = 0.f;
    for (int k = 0; k < KNN; ++k) s += s_red[nn * KNN + k][comp];
    const int bi = bi0 + nn;
    const int c  = comp % 3;
    if (comp < 3) cm_out[bi * 3 + c] = s + cmean[bi * 3 + c];
    else          cv_out[bi * 3 + c] = s + cvar[bi * 3 + c];
  }
}

// ---------------- node MLP: 32 rows per block, residual ----------------
__global__ void __launch_bounds__(256) egnn_node(const float* __restrict__ feats,
                                                 const _Float16* __restrict__ wsh,
                                                 const float* __restrict__ mibuf,
                                                 float* __restrict__ node_out) {
  const int rowStart = blockIdx.x * 32;
  const int tid = threadIdx.x;
  const int wv  = tid >> 5;
  const _Float16* Wn1t = wsh + OFF_WN1T;
  const _Float16* Wn2t = wsh + OFF_WN2T;
  const _Float16* bn1p = wsh + OFF_BN1;
  const _Float16* bn2p = wsh + OFF_BN2;
  const _Float16* featsh = wsh + OFF_FEATSH;

  __shared__ _Float16 s_nin[32 * NIN];   // 12 KB
  __shared__ _Float16 s_nh[32 * NH];     // 16 KB

  for (int t = tid; t < 32 * NIN; t += 256) {
    const int r = t / NIN, c = t - r * NIN;
    const int row = rowStart + r;
    s_nin[t] = (c < DIM) ? featsh[(size_t)row * DIM + c]
                         : (_Float16)mibuf[(size_t)row * MDIM + (c - DIM)];
  }
  __syncthreads();

  // [32x192] @ [192x256], SiLU   (col-strip, 2 row tiles)
  for (int ct = wv; ct < NH / 16; ct += 8) {
    const float bv = (float)bn1p[ct * 16 + (tid & 15)];
    v8f acc[2];
    gemm_colstrip<NIN / 32, 2>(s_nin, NIN, Wn1t, NIN, ct * 16, bv, acc);
#pragma unroll
    for (int rt = 0; rt < 2; ++rt) store_silu(acc[rt], s_nh, NH, rt * 16, ct * 16);
  }
  __syncthreads();

  // [32x256] @ [256x128] + feats residual -> f32 out
  for (int ct = wv; ct < DIM / 16; ct += 8) {
    const float bv = (float)bn2p[ct * 16 + (tid & 15)];
    v8f acc[2];
    gemm_colstrip<NH / 32, 2>(s_nh, NH, Wn2t, NH, ct * 16, bv, acc);
    const int l = tid & 31;
    const int col = ct * 16 + (l & 15);
#pragma unroll
    for (int rt = 0; rt < 2; ++rt) {
      const int r0 = rt * 16 + ((l >> 4) << 3);
#pragma unroll
      for (int r = 0; r < 8; ++r) {
        const int row = rowStart + r0 + r;
        node_out[(size_t)row * DIM + col] = acc[rt][r] + feats[(size_t)row * DIM + col];
      }
    }
  }
}

// ---------------- host launcher ----------------
extern "C" void kernel_launch(void* const* d_in, const int* in_sizes, int n_in,
                              void* d_out, int out_size, void* d_ws, size_t ws_size,
                              hipStream_t stream) {
  (void)in_sizes; (void)n_in; (void)out_size; (void)ws_size;
  const float* feats = (const float*)d_in[0];
  const float* cmean = (const float*)d_in[1];
  const float* cvar  = (const float*)d_in[2];
  const float* We1 = (const float*)d_in[3];
  const float* be1 = (const float*)d_in[4];
  const float* We2 = (const float*)d_in[5];
  const float* be2 = (const float*)d_in[6];
  const float* Wm1 = (const float*)d_in[7];
  const float* bm1 = (const float*)d_in[8];
  const float* Wm2 = (const float*)d_in[9];
  const float* bm2 = (const float*)d_in[10];
  const float* Wv1 = (const float*)d_in[11];
  const float* bv1 = (const float*)d_in[12];
  const float* Wv2 = (const float*)d_in[13];
  const float* bv2 = (const float*)d_in[14];
  const float* Wn1 = (const float*)d_in[15];
  const float* bn1 = (const float*)d_in[16];
  const float* Wn2 = (const float*)d_in[17];
  const float* bn2 = (const float*)d_in[18];

  _Float16* wsh = (_Float16*)d_ws;
  const size_t half_bytes = (HALF_TOTAL * sizeof(_Float16) + 255) & ~(size_t)255;
  const size_t idx_bytes  = (((size_t)Bsz * Npt * KNN * sizeof(int)) + 255) & ~(size_t)255;
  int*   idxbuf = (int*)((char*)d_ws + half_bytes);
  float* mibuf  = (float*)((char*)d_ws + half_bytes + idx_bytes);

  float* node_out = (float*)d_out;
  float* cm_out = node_out + (size_t)Bsz * Npt * DIM;
  float* cv_out = cm_out + (size_t)Bsz * Npt * 3;

  egnn_convert<<<512, 256, 0, stream>>>(feats, We1, be1, We2, be2, Wm1, bm1, Wm2, bm2,
                                        Wv1, bv1, Wv2, bv2, Wn1, bn1, Wn2, bn2, wsh);
  egnn_knn<<<Bsz * Npt, 256, 0, stream>>>(cmean, cvar, idxbuf);
  egnn_edge<<<(Bsz * Npt) / NPB, 256, 0, stream>>>(cmean, cvar, wsh, idxbuf, mibuf,
                                                   cm_out, cv_out);
  egnn_node<<<(Bsz * Npt) / 32, 256, 0, stream>>>(feats, wsh, mibuf, node_out);
}
